// AVPViT_33165737460066
// MI455X (gfx1250) — matrix operand order repaired and checked by reading, not compiled, not run on hardware
//
#include <hip/hip_runtime.h>
#include <hip/hip_bf16.h>
#include <math.h>

// ---------------------------------------------------------------------------
// Problem constants (from reference)
// ---------------------------------------------------------------------------
namespace {
constexpr int Ll   = 8;     // layers
constexpr int Dm   = 768;   // model dim
constexpr int NHh  = 12;    // heads
constexpr int HDd  = 64;    // head dim
constexpr int GGc  = 7;     // local grid
constexpr int SGc  = 16;    // scene grid
constexpr int NLoc = GGc * GGc;   // 49 local tokens
constexpr int NSc  = SGc * SGc;   // 256 scene tokens
constexpr int Bb   = 64;    // batch
constexpr int MLr  = Bb * NLoc;   // 3136 local rows
constexpr int MSr  = Bb * NSc;    // 16384 scene rows
constexpr float SCALEc = 0.125f;  // 1/sqrt(64)
constexpr int SLP  = 64;    // padded leading dim for 49-wide score rows
}

typedef __attribute__((ext_vector_type(16))) _Float16 v16h;
typedef __attribute__((ext_vector_type(8)))  _Float16 v8h;
typedef __attribute__((ext_vector_type(8)))  float    v8f;

static __device__ __forceinline__ v8h cvt8(const float4 f0, const float4 f1)
{
    v8h hv;
    hv[0] = (_Float16)f0.x; hv[1] = (_Float16)f0.y;
    hv[2] = (_Float16)f0.z; hv[3] = (_Float16)f0.w;
    hv[4] = (_Float16)f1.x; hv[5] = (_Float16)f1.y;
    hv[6] = (_Float16)f1.z; hv[7] = (_Float16)f1.w;
    return hv;
}

// ---------------------------------------------------------------------------
// Generic batched GEMM:  C = alpha * (A x B) + bias
//   A: fp32, M x K, row stride lda (mult of 4), (b,h) offset b*sAb+h*sAh
//   B: fp32, K x N (transB=0) or N x K (transB=1), stride ldb (mult of 4)
//   C: fp32, M x N, row stride ldc
// f16 WMMA; fp32->f16 conversion fused into vectorized (b128) loads.
// Tile BM x 64, BK=32, NT=BM threads, NW=BM/32 waves.
// Each wave owns a 32 x 64 strip: 2 A fragments x 4 B fragments = 8 WMMAs
// per k-step (B fragments reused from registers).
// Double-buffered LDS with register staging: next tile's global loads are
// issued before this tile's WMMAs; one barrier per k-step.
// ---------------------------------------------------------------------------
template<int BM>
__global__ __launch_bounds__(BM) void gemm_wmma_kernel(
    const float* __restrict__ A, const float* __restrict__ Bm,
    const float* __restrict__ bias, float* __restrict__ C,
    int M, int N, int K, int lda, int ldb, int ldc,
    size_t sAb, size_t sAh, size_t sBb, size_t sBh, size_t sCb, size_t sCh,
    float alpha, int transB)
{
    constexpr int NT  = BM;        // threads
    constexpr int CPR = NT / 64;   // transB: k-chunks per n row (1 or 2)
    constexpr int BH  = 32 / CPR;  // transB: halves per thread (32 or 16)
    constexpr int BHV = BH / 8;    // transB: v8h per thread (4 or 2)
    constexpr int GRP = NT / 32;   // normal: thread groups (2 or 4)
    constexpr int GN  = 64 / GRP;  // normal: n per thread (32 or 16)

    __shared__ __align__(16) _Float16 As[2][BM][40];
    __shared__ __align__(16) _Float16 Bs[2][64][40];

    const int z  = blockIdx.z;
    const int zb = z / NHh, zh = z % NHh;
    A  += zb * sAb + zh * sAh;
    Bm += zb * sBb + zh * sBh;
    C  += zb * sCb + zh * sCh;

    const int tid  = threadIdx.x;
    const int wave = tid >> 5;
    const int lane = tid & 31;
    const int m0 = blockIdx.y * BM;
    const int n0 = blockIdx.x * 64;

    // ---------------- register staging for one tile ------------------------
    v8h      stA[4];        // one full A row (32 halves)
    v8h      stBt[4];       // transB: BHV used
    _Float16 stBn[32];      // normal: GN used

    auto stageTile = [&](int k0) {
        const bool kfull = (k0 + 32 <= K);
        // ---- A row: r = tid ----
        {
            const int gm = m0 + tid;
            if (kfull && gm < M) {
                const float* src = A + (size_t)gm * lda + k0;
                #pragma unroll
                for (int q = 0; q < 4; ++q) {
                    float4 f0 = *(const float4*)(src + 8 * q);
                    float4 f1 = *(const float4*)(src + 8 * q + 4);
                    stA[q] = cvt8(f0, f1);
                }
            } else {
                const int gmc = (gm < M) ? gm : 0;
                const float* srow = A + (size_t)gmc * lda;
                #pragma unroll
                for (int q = 0; q < 4; ++q)
                    #pragma unroll
                    for (int j = 0; j < 8; ++j) {
                        const int gk  = k0 + 8 * q + j;
                        const int gkc = (gk < K) ? gk : 0;
                        float v = srow[gkc];
                        if (gm >= M || gk >= K) v = 0.0f;
                        stA[q][j] = (_Float16)v;
                    }
            }
        }
        // ---- B tile ----
        if (transB) {
            const int n  = tid / CPR;
            const int kh = (tid % CPR) * BH;
            const int gn = n0 + n;
            if (kfull && gn < N) {
                const float* src = Bm + (size_t)gn * ldb + (k0 + kh);
                #pragma unroll
                for (int q = 0; q < BHV; ++q) {
                    float4 f0 = *(const float4*)(src + 8 * q);
                    float4 f1 = *(const float4*)(src + 8 * q + 4);
                    stBt[q] = cvt8(f0, f1);
                }
            } else {
                const int gnc = (gn < N) ? gn : 0;
                const float* srow = Bm + (size_t)gnc * ldb;
                #pragma unroll
                for (int q = 0; q < BHV; ++q)
                    #pragma unroll
                    for (int j = 0; j < 8; ++j) {
                        const int gk  = k0 + kh + 8 * q + j;
                        const int gkc = (gk < K) ? gk : 0;
                        float v = srow[gkc];
                        if (gn >= N || gk >= K) v = 0.0f;
                        stBt[q][j] = (_Float16)v;
                    }
            }
        } else {
            const int kk = tid & 31;
            const int ng = (tid >> 5) * GN;
            const int gk = k0 + kk;
            const bool kok = (gk < K);
            const float* src = Bm + (size_t)(kok ? gk : 0) * ldb + n0 + ng;
            #pragma unroll
            for (int q = 0; q < GN / 4; ++q) {
                const int gn = n0 + ng + 4 * q;
                if (kok && gn + 3 < N) {
                    float4 f = *(const float4*)(src + 4 * q);
                    stBn[4 * q + 0] = (_Float16)f.x;
                    stBn[4 * q + 1] = (_Float16)f.y;
                    stBn[4 * q + 2] = (_Float16)f.z;
                    stBn[4 * q + 3] = (_Float16)f.w;
                } else {
                    #pragma unroll
                    for (int jj = 0; jj < 4; ++jj) {
                        const int gnn = gn + jj;
                        const float v = (kok && gnn < N) ? src[4 * q + jj] : 0.0f;
                        stBn[4 * q + jj] = (_Float16)v;
                    }
                }
            }
        }
    };

    auto commitTile = [&](int buf) {
        #pragma unroll
        for (int q = 0; q < 4; ++q)
            *(v8h*)&As[buf][tid][8 * q] = stA[q];
        if (transB) {
            const int n  = tid / CPR;
            const int kh = (tid % CPR) * BH;
            #pragma unroll
            for (int q = 0; q < BHV; ++q)
                *(v8h*)&Bs[buf][n][kh + 8 * q] = stBt[q];
        } else {
            const int kk = tid & 31;
            const int ng = (tid >> 5) * GN;
            #pragma unroll
            for (int j = 0; j < GN; ++j)
                Bs[buf][ng + j][kk] = stBn[j];
        }
    };

    v8f acc[2][4] = {};

    // ---------------- pipelined main loop ----------------------------------
    stageTile(0);
    commitTile(0);
    __syncthreads();

    int cur = 0;
    for (int k0 = 0; k0 < K; k0 += 32) {
        const bool more = (k0 + 32) < K;
        if (more) stageTile(k0 + 32);     // global loads fly under the WMMAs

        // ---- A fragments (two 16-row strips per wave) ----
        const int kbA = (lane < 16) ? 0 : 8;
        v16h afrag[2];
        #pragma unroll
        for (int h = 0; h < 2; ++h) {
            const int mrow = wave * 32 + h * 16 + (lane & 15);
            v8h lo = *(const v8h*)&As[cur][mrow][kbA];
            v8h hi = *(const v8h*)&As[cur][mrow][16 + kbA];
            afrag[h] = __builtin_shufflevector(lo, hi,
                       0,1,2,3,4,5,6,7,8,9,10,11,12,13,14,15);
        }
        // ---- 4 B fragments, each reused for both A strips ----
        const int kbB = (lane < 16) ? 0 : 8;
        const int nl  = lane & 15;
        #pragma unroll
        for (int c = 0; c < 4; ++c) {
            const int n = c * 16 + nl;
            v8h bLo = *(const v8h*)&Bs[cur][n][kbB];
            v8h bHi = *(const v8h*)&Bs[cur][n][kbB + 16];
            v16h bf = __builtin_shufflevector(bLo, bHi,
                      0,1,2,3,4,5,6,7,8,9,10,11,12,13,14,15);
            acc[0][c] = __builtin_amdgcn_wmma_f32_16x16x32_f16(
                            false, afrag[0], false, bf, (short)0, acc[0][c],
                            false, false);
            acc[1][c] = __builtin_amdgcn_wmma_f32_16x16x32_f16(
                            false, afrag[1], false, bf, (short)0, acc[1][c],
                            false, false);
        }

        if (more) commitTile(cur ^ 1);
        __syncthreads();
        cur ^= 1;
    }

    // ---- epilogue ----------------------------------------------------------
    const int nl = lane & 15;
    #pragma unroll
    for (int h = 0; h < 2; ++h) {
        const int mb = m0 + wave * 32 + h * 16 + (lane >> 4) * 8;
        #pragma unroll
        for (int c = 0; c < 4; ++c) {
            const int gn = n0 + c * 16 + nl;
            if (gn < N) {
                const float bv = bias ? bias[gn] : 0.0f;
                #pragma unroll
                for (int r = 0; r < 8; ++r) {
                    const int gm = mb + r;
                    if (gm < M) C[(size_t)gm * ldc + gn] = alpha * acc[h][c][r] + bv;
                }
            }
        }
    }
}

// ---------------------------------------------------------------------------
// RoPE, in-place on [rows, ldx] at head layout (h*64 + d). One block per row.
// ---------------------------------------------------------------------------
__global__ __launch_bounds__(128) void rope_kernel(
    float* __restrict__ X, int ldx,
    const float* __restrict__ centers, const float* __restrict__ scales,
    const float* __restrict__ periods, int T, int G, int isLocal)
{
    const int row = blockIdx.x;
    const int b = row / T, t = row % T;
    const float cx = ((t % G) + 0.5f) / (float)G;
    const float cy = ((t / G) + 0.5f) / (float)G;
    float px, py;
    if (isLocal) {
        px = centers[2 * b + 0] + scales[2 * b + 0] * (cx - 0.5f);
        py = centers[2 * b + 1] + scales[2 * b + 1] * (cy - 0.5f);
    } else { px = cx; py = cy; }
    float* xr = X + (size_t)row * ldx;
    for (int e = threadIdx.x; e < NHh * 32; e += blockDim.x) {
        const int h = e >> 5, j = e & 31;
        const float pos = (j < 16) ? px : py;      // j = comp*16 + per
        const float ang = 6.283185307179586f * pos / periods[j & 15];
        const float cs = cosf(ang), sn = sinf(ang);
        const int i1 = h * 64 + j, i2 = i1 + 32;
        const float x1 = xr[i1], x2 = xr[i2];
        xr[i1] = x1 * cs - x2 * sn;
        xr[i2] = x1 * sn + x2 * cs;
    }
}

// ---------------------------------------------------------------------------
// LayerNorm over D=768 (one block of 256 per row, 3 elems/thread)
// ---------------------------------------------------------------------------
__global__ __launch_bounds__(256) void ln_kernel(
    const float* __restrict__ X, const float* __restrict__ w,
    const float* __restrict__ bwt, float* __restrict__ Y)
{
    const int row = blockIdx.x;
    const float* xr = X + (size_t)row * Dm;
    float* yr = Y + (size_t)row * Dm;
    __shared__ float rs[8], rs2[8];
    float s = 0.f, s2 = 0.f, vals[3];
    #pragma unroll
    for (int i = 0; i < 3; ++i) {
        const float v = xr[threadIdx.x + 256 * i];
        vals[i] = v; s += v; s2 += v * v;
    }
    for (int off = 16; off; off >>= 1) {
        s  += __shfl_down(s,  off, 32);
        s2 += __shfl_down(s2, off, 32);
    }
    const int lane = threadIdx.x & 31, wv = threadIdx.x >> 5;
    if (!lane) { rs[wv] = s; rs2[wv] = s2; }
    __syncthreads();
    if (threadIdx.x == 0) {
        float a = 0.f, c = 0.f;
        for (int i = 0; i < 8; ++i) { a += rs[i]; c += rs2[i]; }
        rs[0] = a; rs2[0] = c;
    }
    __syncthreads();
    const float mean = rs[0] * (1.0f / Dm);
    const float var  = rs2[0] * (1.0f / Dm) - mean * mean;
    const float inv  = rsqrtf(var + 1e-6f);
    #pragma unroll
    for (int i = 0; i < 3; ++i) {
        const int e = threadIdx.x + 256 * i;
        yr[e] = (vals[i] - mean) * inv * w[e] + bwt[e];
    }
}

// ---------------------------------------------------------------------------
// Row softmax, in-place. Row length m, padded row stride ld.
// ---------------------------------------------------------------------------
__global__ __launch_bounds__(256) void softmax_kernel(float* __restrict__ S,
                                                      int m, int ld)
{
    float* sr = S + (size_t)blockIdx.x * ld;
    __shared__ float red[8];
    const int tid = threadIdx.x, lane = tid & 31, wv = tid >> 5;
    float mx = -3.4e38f;
    for (int e = tid; e < m; e += 256) mx = fmaxf(mx, sr[e]);
    for (int off = 16; off; off >>= 1) mx = fmaxf(mx, __shfl_down(mx, off, 32));
    if (!lane) red[wv] = mx;
    __syncthreads();
    if (tid == 0) {
        float a = red[0];
        for (int i = 1; i < 8; ++i) a = fmaxf(a, red[i]);
        red[0] = a;
    }
    __syncthreads();
    mx = red[0];
    __syncthreads();
    float sum = 0.f;
    for (int e = tid; e < m; e += 256) {
        const float v = __expf(sr[e] - mx);
        sr[e] = v; sum += v;
    }
    for (int off = 16; off; off >>= 1) sum += __shfl_down(sum, off, 32);
    if (!lane) red[wv] = sum;
    __syncthreads();
    if (tid == 0) {
        float a = 0.f;
        for (int i = 0; i < 8; ++i) a += red[i];
        red[0] = a;
    }
    __syncthreads();
    const float inv = 1.0f / red[0];
    for (int e = tid; e < m; e += 256) sr[e] *= inv;
}

// ---------------------------------------------------------------------------
// Exact GELU (erf-based), in-place
// ---------------------------------------------------------------------------
__global__ void gelu_kernel(float* __restrict__ X, size_t n)
{
    size_t i = (size_t)blockIdx.x * blockDim.x + threadIdx.x;
    const size_t stride = (size_t)gridDim.x * blockDim.x;
    for (; i < n; i += stride) {
        const float x = X[i];
        X[i] = 0.5f * x * (1.0f + erff(x * 0.70710678118654752f));
    }
}

// ---------------------------------------------------------------------------
// X += gate[col] * Y  (gate == null -> plain residual add)
// ---------------------------------------------------------------------------
__global__ void gate_residual_kernel(float* __restrict__ X, const float* __restrict__ Y,
                                     const float* __restrict__ gate, size_t n)
{
    size_t i = (size_t)blockIdx.x * blockDim.x + threadIdx.x;
    const size_t stride = (size_t)gridDim.x * blockDim.x;
    for (; i < n; i += stride) {
        const float g = gate ? gate[i % Dm] : 1.0f;
        X[i] += g * Y[i];
    }
}

// ---------------------------------------------------------------------------
// dst[i] = src[i % per]  (broadcast scene tokens over batch)
// ---------------------------------------------------------------------------
__global__ void broadcast_kernel(float* __restrict__ dst, const float* __restrict__ src,
                                 size_t per, size_t n)
{
    size_t i = (size_t)blockIdx.x * blockDim.x + threadIdx.x;
    const size_t stride = (size_t)gridDim.x * blockDim.x;
    for (; i < n; i += stride) dst[i] = src[i % per];
}

// ---------------------------------------------------------------------------
// Host-side launch helpers
// ---------------------------------------------------------------------------
static inline void launch_gemm(hipStream_t s,
    const float* A, const float* B, const float* bias, float* C,
    int M, int N, int K, int lda, int ldb, int ldc,
    size_t sAb, size_t sAh, size_t sBb, size_t sBh, size_t sCb, size_t sCh,
    float alpha, int transB, int nbh)
{
    if (M >= 128) {
        dim3 grid((N + 63) / 64, (M + 127) / 128, nbh);
        gemm_wmma_kernel<128><<<grid, 128, 0, s>>>(A, B, bias, C, M, N, K,
            lda, ldb, ldc, sAb, sAh, sBb, sBh, sCb, sCh, alpha, transB);
    } else {
        dim3 grid((N + 63) / 64, (M + 63) / 64, nbh);
        gemm_wmma_kernel<64><<<grid, 64, 0, s>>>(A, B, bias, C, M, N, K,
            lda, ldb, ldc, sAb, sAh, sBb, sBh, sCb, sCh, alpha, transB);
    }
}

// Plain (non-batched) projection: lda=K, ldb=N, ldc=N
static inline void launch_proj(hipStream_t s, const float* A, const float* B,
                               const float* bias, float* C, int M, int N, int K)
{
    launch_gemm(s, A, B, bias, C, M, N, K, K, N, N, 0, 0, 0, 0, 0, 0, 1.0f, 0, 1);
}

extern "C" void kernel_launch(void* const* d_in, const int* in_sizes, int n_in,
                              void* d_out, int out_size, void* d_ws, size_t ws_size,
                              hipStream_t stream)
{
    (void)in_sizes; (void)n_in; (void)out_size; (void)ws_size;

    const float* in_local   = (const float*)d_in[0];
    const float* centers    = (const float*)d_in[1];
    const float* scales     = (const float*)d_in[2];
    const float* scene_tok  = (const float*)d_in[3];
    const float* periods    = (const float*)d_in[4];
    const float* read_gate  = (const float*)d_in[5];
    const float* write_gate = (const float*)d_in[6];
    const float* ln1w = (const float*)d_in[7];
    const float* ln1b = (const float*)d_in[8];
    const float* ln2w = (const float*)d_in[9];
    const float* ln2b = (const float*)d_in[10];
    const float* Wqkv = (const float*)d_in[11];
    const float* bqkv = (const float*)d_in[12];
    const float* Wp   = (const float*)d_in[13];
    const float* bp   = (const float*)d_in[14];
    const float* W1   = (const float*)d_in[15];
    const float* b1   = (const float*)d_in[16];
    const float* W2   = (const float*)d_in[17];
    const float* b2   = (const float*)d_in[18];
    // order: rq, rk, rv, ro, wq, wk, wv, wo
    const float* AW[8]; const float* Ab[8];
    for (int j = 0; j < 8; ++j) {
        AW[j] = (const float*)d_in[19 + 2 * j];
        Ab[j] = (const float*)d_in[20 + 2 * j];
    }

    // -------- workspace layout (fp32), buffers aliased across phases -------
    float* ws = (float*)d_ws;
    size_t off = 0;
    float* local = ws + off; off += (size_t)MLr * Dm;            // 3136 x 768
    float* scene = ws + off; off += (size_t)MSr * Dm;            // 16384 x 768
    float* Qb    = ws + off; off += (size_t)MSr * Dm;            // also QKV (3136 x 2304)
    float* Kb    = ws + off; off += (size_t)MSr * Dm;            // also H1 (3136 x 3072)
    float* Vb    = ws + off; off += (size_t)MSr * Dm;            // also LN output
    float* Sb    = ws + off; off += (size_t)Bb * NHh * NSc * SLP;  // padded scores
    float* Ob    = ws + off; off += (size_t)MSr * Dm;            // merged attn out
    float* Tb    = ws + off; off += (size_t)MSr * Dm;            // proj temp

    const size_t DD = (size_t)Dm * Dm;

    // -------- init: local <- input, scene <- broadcast(scene_tokens) ------
    hipMemcpyAsync(local, in_local, sizeof(float) * (size_t)MLr * Dm,
                   hipMemcpyDeviceToDevice, stream);
    broadcast_kernel<<<4096, 256, 0, stream>>>(scene, scene_tok,
                                               (size_t)NSc * Dm, (size_t)MSr * Dm);

    const int nbh = Bb * NHh; // 768 (b,h) pairs

    for (int i = 0; i < Ll; ++i) {
        // ================= READ cross-attn: local <- attend(scene) ========
        launch_proj(stream, local, AW[0] + i * DD, Ab[0] + (size_t)i * Dm, Qb, MLr, Dm, Dm);
        launch_proj(stream, scene, AW[1] + i * DD, Ab[1] + (size_t)i * Dm, Kb, MSr, Dm, Dm);
        launch_proj(stream, scene, AW[2] + i * DD, Ab[2] + (size_t)i * Dm, Vb, MSr, Dm, Dm);
        rope_kernel<<<MLr, 128, 0, stream>>>(Qb, Dm, centers, scales, periods, NLoc, GGc, 1);
        rope_kernel<<<MSr, 128, 0, stream>>>(Kb, Dm, centers, scales, periods, NSc, SGc, 0);
        // S[b,h,n,m] = scale * Q Kt   (m = 256, natural ld)
        launch_gemm(stream, Qb, Kb, nullptr, Sb, NLoc, NSc, HDd, Dm, Dm, NSc,
                    (size_t)NLoc * Dm, HDd, (size_t)NSc * Dm, HDd,
                    (size_t)NHh * NLoc * NSc, (size_t)NLoc * NSc,
                    SCALEc, 1, nbh);
        softmax_kernel<<<Bb * NHh * NLoc, 256, 0, stream>>>(Sb, NSc, NSc);
        // O = P V  (merged head layout, ldc = D)
        launch_gemm(stream, Sb, Vb, nullptr, Ob, NLoc, HDd, NSc, NSc, Dm, Dm,
                    (size_t)NHh * NLoc * NSc, (size_t)NLoc * NSc,
                    (size_t)NSc * Dm, HDd,
                    (size_t)NLoc * Dm, HDd,
                    1.0f, 0, nbh);
        launch_proj(stream, Ob, AW[3] + i * DD, Ab[3] + (size_t)i * Dm, Tb, MLr, Dm, Dm);
        gate_residual_kernel<<<4096, 256, 0, stream>>>(local, Tb,
                               read_gate + (size_t)i * Dm, (size_t)MLr * Dm);

        // ================= Backend block (local self-attn + MLP) ==========
        float* Hb   = Vb;  // LN output, 3136 x 768
        float* QKVb = Qb;  // 3136 x 2304
        float* H1b  = Kb;  // 3136 x 3072
        ln_kernel<<<MLr, 256, 0, stream>>>(local, ln1w + (size_t)i * Dm,
                                           ln1b + (size_t)i * Dm, Hb);
        launch_proj(stream, Hb, Wqkv + (size_t)i * Dm * 3 * Dm,
                    bqkv + (size_t)i * 3 * Dm, QKVb, MLr, 3 * Dm, Dm);
        rope_kernel<<<MLr, 128, 0, stream>>>(QKVb,      3 * Dm, centers, scales, periods, NLoc, GGc, 1);
        rope_kernel<<<MLr, 128, 0, stream>>>(QKVb + Dm, 3 * Dm, centers, scales, periods, NLoc, GGc, 1);
        // S2: 49 x 49 scores, padded row stride SLP=64
        launch_gemm(stream, QKVb, QKVb + Dm, nullptr, Sb, NLoc, NLoc, HDd,
                    3 * Dm, 3 * Dm, SLP,
                    (size_t)NLoc * 3 * Dm, HDd, (size_t)NLoc * 3 * Dm, HDd,
                    (size_t)NHh * NLoc * SLP, (size_t)NLoc * SLP,
                    SCALEc, 1, nbh);
        softmax_kernel<<<Bb * NHh * NLoc, 256, 0, stream>>>(Sb, NLoc, SLP);
        launch_gemm(stream, Sb, QKVb + 2 * Dm, nullptr, Ob, NLoc, HDd, NLoc,
                    SLP, 3 * Dm, Dm,
                    (size_t)NHh * NLoc * SLP, (size_t)NLoc * SLP,
                    (size_t)NLoc * 3 * Dm, HDd,
                    (size_t)NLoc * Dm, HDd,
                    1.0f, 0, nbh);
        launch_proj(stream, Ob, Wp + i * DD, bp + (size_t)i * Dm, Tb, MLr, Dm, Dm);
        gate_residual_kernel<<<4096, 256, 0, stream>>>(local, Tb, nullptr,
                                                       (size_t)MLr * Dm);
        ln_kernel<<<MLr, 256, 0, stream>>>(local, ln2w + (size_t)i * Dm,
                                           ln2b + (size_t)i * Dm, Hb);
        launch_proj(stream, Hb, W1 + (size_t)i * Dm * 4 * Dm,
                    b1 + (size_t)i * 4 * Dm, H1b, MLr, 4 * Dm, Dm);
        gelu_kernel<<<4096, 256, 0, stream>>>(H1b, (size_t)MLr * 4 * Dm);
        launch_proj(stream, H1b, W2 + (size_t)i * 4 * Dm * Dm,
                    b2 + (size_t)i * Dm, Tb, MLr, Dm, 4 * Dm);
        gate_residual_kernel<<<4096, 256, 0, stream>>>(local, Tb, nullptr,
                                                       (size_t)MLr * Dm);

        // ================= WRITE cross-attn: scene <- attend(local) =======
        launch_proj(stream, scene, AW[4] + i * DD, Ab[4] + (size_t)i * Dm, Qb, MSr, Dm, Dm);
        launch_proj(stream, local, AW[5] + i * DD, Ab[5] + (size_t)i * Dm, Kb, MLr, Dm, Dm);
        launch_proj(stream, local, AW[6] + i * DD, Ab[6] + (size_t)i * Dm, Vb, MLr, Dm, Dm);
        rope_kernel<<<MSr, 128, 0, stream>>>(Qb, Dm, centers, scales, periods, NSc, SGc, 0);
        rope_kernel<<<MLr, 128, 0, stream>>>(Kb, Dm, centers, scales, periods, NLoc, GGc, 1);
        // S: 256 x 49 scores, padded row stride SLP=64
        launch_gemm(stream, Qb, Kb, nullptr, Sb, NSc, NLoc, HDd, Dm, Dm, SLP,
                    (size_t)NSc * Dm, HDd, (size_t)NLoc * Dm, HDd,
                    (size_t)NHh * NSc * SLP, (size_t)NSc * SLP,
                    SCALEc, 1, nbh);
        softmax_kernel<<<Bb * NHh * NSc, 256, 0, stream>>>(Sb, NLoc, SLP);
        launch_gemm(stream, Sb, Vb, nullptr, Ob, NSc, HDd, NLoc, SLP, Dm, Dm,
                    (size_t)NHh * NSc * SLP, (size_t)NSc * SLP,
                    (size_t)NLoc * Dm, HDd,
                    (size_t)NSc * Dm, HDd,
                    1.0f, 0, nbh);
        launch_proj(stream, Ob, AW[7] + i * DD, Ab[7] + (size_t)i * Dm, Tb, MSr, Dm, Dm);
        gate_residual_kernel<<<8192, 256, 0, stream>>>(scene, Tb,
                               write_gate + (size_t)i * Dm, (size_t)MSr * Dm);
    }

    // -------- outputs: (local, scene) concatenated ------------------------
    float* outp = (float*)d_out;
    hipMemcpyAsync(outp, local, sizeof(float) * (size_t)MLr * Dm,
                   hipMemcpyDeviceToDevice, stream);
    hipMemcpyAsync(outp + (size_t)MLr * Dm, scene,
                   sizeof(float) * (size_t)MSr * Dm,
                   hipMemcpyDeviceToDevice, stream);
}
